// OutputModule_22737556865607
// MI455X (gfx1250) — matrix operand order repaired and checked by reading, not compiled, not run on hardware
//
#include <hip/hip_runtime.h>
#include <cstdint>
#include <cstddef>

#define NN     50000
#define NE     262144
#define EMBED  512
#define DIN    1536
#define FF     1024
#define NG     64

typedef __attribute__((ext_vector_type(4)))  unsigned int u32x4;
typedef __attribute__((ext_vector_type(8)))  int          i32x8;
typedef __attribute__((ext_vector_type(4)))  int          i32x4;
typedef __attribute__((ext_vector_type(8)))  float        v8f;
typedef __attribute__((ext_vector_type(16))) __bf16       v16bf;
typedef __attribute__((ext_vector_type(8)))  __bf16       v8bf;
typedef __attribute__((ext_vector_type(4)))  __bf16       v4bf;

// LDS layout (bytes): A panel | B double buffer | idx | scalar partials | energy partials
#define A_BYTES   (64 * 1536 * 2)        // 196608
#define B_BYTES   (2 * 512 * 32 * 2)     // 65536 (two 32K buffers)
#define LDS_TOTAL (A_BYTES + B_BYTES + 64*4*2 + 256*4 + 64*4)  // 263936 < 320KB

// ---------------------------------------------------------------------------
// TDM: async 2D tile load (512 rows of Wᵀ × 32 K-elements, bf16) into LDS.
// D# packing per CDNA5 ISA.
// ---------------------------------------------------------------------------
__device__ __forceinline__ void tdm_issue_b(const __bf16* gsrc, unsigned lds_byte_addr) {
  unsigned long long va = (unsigned long long)(uintptr_t)gsrc;
  u32x4 g0;
  g0[0] = 1u;                                   // count = 1 valid descriptor
  g0[1] = lds_byte_addr;                        // lds_addr
  g0[2] = (unsigned)(va & 0xffffffffu);         // global_addr[31:0]
  g0[3] = (unsigned)((va >> 32) & 0x01ffffffu)  // global_addr[56:32]
        | (2u << 30);                           // type = 2 ("image")
  i32x8 g1;
  g1[0] = 1 << 16;             // data_size = 1 (2-byte elements)
  g1[1] = (int)(1536u << 16);  // tensor_dim0[15:0] = 1536 (K length)
  g1[2] = (int)(1024u << 16);  // tensor_dim0 hi = 0 | tensor_dim1[15:0] = 1024
  g1[3] = (int)(32u << 16);    // tensor_dim1 hi = 0 | tile_dim0 = 32
  g1[4] = 512;                 // tile_dim1 = 512 | tile_dim2 = 0
  g1[5] = 1536;                // tensor_dim0_stride[31:0] = 1536 elements
  g1[6] = 0;                   // stride hi | tensor_dim1_stride lo = 0
  g1[7] = 0;
  i32x4 z4 = {0, 0, 0, 0};
#if __has_include(<hip/amd_detail/amd_gfx1250_TDM.h>)
  i32x8 z8 = {0, 0, 0, 0, 0, 0, 0, 0};
  __builtin_amdgcn_tensor_load_to_lds(g0, g1, z4, z4, z8, 0);
#else
  __builtin_amdgcn_tensor_load_to_lds(g0, g1, z4, z4, 0);
#endif
}

// Hardware tanh (V_TANH_F32 trans op on gfx1250) or v_exp_f32-based fallback;
// both avoid the ~25-op OCML tanhf expansion that dominated the VALU count.
__device__ __forceinline__ float fast_tanh(float u) {
#if __has_builtin(__builtin_amdgcn_tanhf)
  return __builtin_amdgcn_tanhf(u);
#else
  // tanh(u) = 1 - 2/(1+exp(2u)); saturates correctly at +/-1 for large |u|
  return 1.0f - 2.0f / (1.0f + __expf(2.0f * u));
#endif
}

__device__ __forceinline__ float gelu_tanh(float v) {
  float v3 = v * v * v;
  return 0.5f * v * (1.0f + fast_tanh(0.79788456080286535588f * (v + 0.044715f * v3)));
}

// ---------------------------------------------------------------------------
// Main fused kernel: 64 edges/block, bf16 WMMA over K=1536, N=2048 (both MLPs)
// ---------------------------------------------------------------------------
extern "C" __global__ __launch_bounds__(256, 1)
void edge_mlp_kernel(const float* __restrict__ x, const float* __restrict__ pos,
                     const float* __restrict__ b1e, const float* __restrict__ W2e,
                     const float* __restrict__ b2e,
                     const float* __restrict__ b1f, const float* __restrict__ W2f,
                     const float* __restrict__ b2f,
                     const int* __restrict__ batch, const int* __restrict__ edge_index,
                     const __bf16* __restrict__ W1eT, const __bf16* __restrict__ W1fT,
                     float* __restrict__ forces, float* __restrict__ ePart) {
  extern __shared__ char smem[];
  __bf16* Abuf = (__bf16*)smem;
  __bf16* Bbuf = (__bf16*)(smem + A_BYTES);
  int*    sSrc = (int*)(smem + A_BYTES + B_BYTES);
  int*    sDst = sSrc + 64;
  float*  sE   = (float*)(sDst + 64);  // [2][64] N-half partials
  float*  sF   = sE + 128;             // [2][64]
  float*  eG   = sF + 128;             // [64] per-graph energy partial

  const int t    = threadIdx.x;
  const int lane = t & 31;
  const int w    = t >> 5;
  const unsigned BldsBase = (unsigned)(uintptr_t)(void*)Bbuf;  // LDS byte address

  // ---- indices + zero per-graph partials ----
  if (t < 64) {
    int eg  = blockIdx.x * 64 + t;
    sSrc[t] = edge_index[eg];
    sDst[t] = edge_index[NE + eg];
    eG[t]   = 0.0f;
  }
  __syncthreads();

  // Kick the first TDM B-tile load (chunk 0, k=0) so DMA overlaps the A gather.
  if (t < 32) tdm_issue_b(W1eT, BldsBase);

  // ---- gather A = [x[src] | x[dst] | edge_tok] -> bf16 in LDS ----
  {
    int el = t >> 2, sub = t & 3;                 // 4 threads per edge
    int eg = blockIdx.x * 64 + el;
    const float* base0 = x + (size_t)sSrc[el] * EMBED;
    const float* base1 = x + (size_t)sDst[el] * EMBED;
    const float* base2 = x + (size_t)(NN + eg) * EMBED;
    __bf16* arow = Abuf + (size_t)el * DIN;
#pragma unroll 8
    for (int j = 0; j < 96; ++j) {
      int uu  = sub * 96 + j;
      int seg = uu >> 7;
      int q   = uu & 127;
      const float* bp = (seg == 0) ? base0 : ((seg == 1) ? base1 : base2);
      float4 f = *(const float4*)(bp + q * 4);
      v4bf h;
      h[0] = (__bf16)f.x; h[1] = (__bf16)f.y; h[2] = (__bf16)f.z; h[3] = (__bf16)f.w;
      *(v4bf*)(arow + uu * 4) = h;
    }
  }
  __syncthreads();

  const int m_tile = w >> 1, nhalf = w & 1;
  const int half16 = lane >> 4, l15 = lane & 15;
  const __bf16* Arow = Abuf + (size_t)(m_tile * 16 + l15) * DIN;

  float outE[8], outF[8];
#pragma unroll
  for (int r = 0; r < 8; ++r) { outE[r] = 0.0f; outF[r] = 0.0f; }

  for (int chunk = 0; chunk < 4; ++chunk) {       // chunks 0,1 = energy MLP; 2,3 = force MLP
    const __bf16* WT = (chunk < 2) ? W1eT : W1fT;
    const float*  b1 = (chunk < 2) ? b1e : b1f;
    const float*  w2 = (chunk < 2) ? W2e : W2f;
    const int  nbase = (chunk & 1) * 512;

    v8f acc[16];
#pragma unroll
    for (int nt = 0; nt < 16; ++nt)
#pragma unroll
      for (int r = 0; r < 8; ++r) acc[nt][r] = 0.0f;

    if (chunk > 0 && t < 32)                       // chunk 0 k=0 was preloaded above
      tdm_issue_b(WT + (size_t)nbase * DIN, BldsBase);

    for (int ks = 0; ks < 48; ++ks) {
      if (t < 32) {
        if (ks + 1 < 48) {
          tdm_issue_b(WT + (size_t)nbase * DIN + (ks + 1) * 32,
                      BldsBase + ((unsigned)((ks + 1) & 1)) * 32768u);
          __builtin_amdgcn_s_wait_tensorcnt(1);    // in-order: oldest (this ks) done
        } else {
          __builtin_amdgcn_s_wait_tensorcnt(0);
        }
      }
      __syncthreads();                             // B buffer (ks&1) visible to all waves

      const int k0 = ks * 32;
      // A frag, contiguous K order matching B: lanes 0-15 K{0..15}, lanes 16-31 K{16..31}
      v16bf a  = *(const v16bf*)(Arow + k0 + half16 * 16);

      // B frags: register double-buffer so each ds_load pair has a full WMMA
      // of latency to complete instead of load->wait(0)->wmma serialization.
      const __bf16* Bbase = Bbuf + (ks & 1) * 16384
                          + (size_t)(nhalf * 256 + l15) * 32 + half16 * 16;
      v16bf bcur = *(const v16bf*)Bbase;
#pragma unroll
      for (int nt = 0; nt < 16; ++nt) {
        v16bf bnext = bcur;
        if (nt < 15) bnext = *(const v16bf*)(Bbase + (nt + 1) * 512);  // 16 rows * 32 elems
        acc[nt] = __builtin_amdgcn_wmma_f32_16x16x32_bf16(
            false, a, false, bcur, (short)0, acc[nt], false, false);
        bcur = bnext;
      }
      __syncthreads();                             // consumption done before buffer reuse
    }

    // epilogue: +b1 -> gelu -> dot with W2 column, fold into per-row scalars
#pragma unroll
    for (int nt = 0; nt < 16; ++nt) {
      const int   ncol = nbase + nhalf * 256 + nt * 16 + l15;
      const float bv   = b1[ncol];
      const float wv   = w2[ncol];
#pragma unroll
      for (int r = 0; r < 8; ++r) {
        float g = gelu_tanh(acc[nt][r] + bv);
        if (chunk < 2) outE[r] += g * wv; else outF[r] += g * wv;
      }
    }
  }

  // cross-lane reduce over the 16 N-columns held per half-wave
#pragma unroll
  for (int r = 0; r < 8; ++r) {
    float e = outE[r], f = outF[r];
#pragma unroll
    for (int m = 1; m <= 8; m <<= 1) {
      e += __shfl_xor(e, m, 32);
      f += __shfl_xor(f, m, 32);
    }
    if (l15 == 0) {
      int row = m_tile * 16 + half16 * 8 + r;      // C layout: VGPR r <-> M row
      sE[nhalf * 64 + row] = e;
      sF[nhalf * 64 + row] = f;
    }
  }
  __syncthreads();

  if (t < 64) {
    float se = sE[t] + sE[64 + t] + b2e[0];
    float sf = sF[t] + sF[64 + t] + b2f[0];
    int s = sSrc[t], d = sDst[t];
    float dx = pos[s * 3 + 0] - pos[d * 3 + 0];
    float dy = pos[s * 3 + 1] - pos[d * 3 + 1];
    float dz = pos[s * 3 + 2] - pos[d * 3 + 2];
    float nrm = sqrtf(dx * dx + dy * dy + dz * dz);
    float inv = sf / fmaxf(nrm, 1e-12f);
    atomicAdd(&forces[(size_t)s * 3 + 0], dx * inv);
    atomicAdd(&forces[(size_t)s * 3 + 1], dy * inv);
    atomicAdd(&forces[(size_t)s * 3 + 2], dz * inv);
    atomicAdd(&eG[batch[s]], se);                  // LDS pre-reduction per graph
  }
  __syncthreads();
  if (t < 64) ePart[(size_t)blockIdx.x * 64 + t] = eG[t];
}

// ---------------------------------------------------------------------------
// Prepass: transpose + convert W1 (f32 [1536][1024]) -> bf16 Wᵀ [1024][1536]
// ---------------------------------------------------------------------------
extern "C" __global__ void w1_transpose_cvt(const float* __restrict__ W1e,
                                            const float* __restrict__ W1f,
                                            __bf16* __restrict__ WTe,
                                            __bf16* __restrict__ WTf) {
  __shared__ float tile[32][33];
  const float* W  = blockIdx.z ? W1f : W1e;
  __bf16*      WT = blockIdx.z ? WTf : WTe;
  int k0 = blockIdx.x * 32;
  int n0 = blockIdx.y * 32;
  int tx = threadIdx.x, ty = threadIdx.y;
#pragma unroll
  for (int i = 0; i < 4; ++i)
    tile[ty + 8 * i][tx] = W[(size_t)(k0 + ty + 8 * i) * FF + (n0 + tx)];
  __syncthreads();
#pragma unroll
  for (int i = 0; i < 4; ++i)
    WT[(size_t)(n0 + ty + 8 * i) * DIN + (k0 + tx)] = (__bf16)tile[tx][ty + 8 * i];
}

extern "C" __global__ void zero_out(float* __restrict__ out, int n) {
  int i = blockIdx.x * blockDim.x + threadIdx.x;
  if (i < n) out[i] = 0.0f;
}

extern "C" __global__ void energy_reduce(const float* __restrict__ ePart,
                                         float* __restrict__ energy) {
  int g = blockIdx.x;
  int t = threadIdx.x;
  float s = 0.0f;
  for (int i = t; i < NE / 64; i += 256) s += ePart[(size_t)i * 64 + g];
#pragma unroll
  for (int m = 1; m < 32; m <<= 1) s += __shfl_xor(s, m, 32);
  __shared__ float red[8];
  if ((t & 31) == 0) red[t >> 5] = s;
  __syncthreads();
  if (t == 0) {
    float tot = 0.0f;
#pragma unroll
    for (int i = 0; i < 8; ++i) tot += red[i];
    energy[g] = tot;
  }
}

// ---------------------------------------------------------------------------
extern "C" void kernel_launch(void* const* d_in, const int* in_sizes, int n_in,
                              void* d_out, int out_size, void* d_ws, size_t ws_size,
                              hipStream_t stream) {
  const float* x    = (const float*)d_in[0];
  const float* pos  = (const float*)d_in[1];
  const float* W1e  = (const float*)d_in[2];
  const float* b1e  = (const float*)d_in[3];
  const float* W2e  = (const float*)d_in[4];
  const float* b2e  = (const float*)d_in[5];
  const float* W1f  = (const float*)d_in[6];
  const float* b1f  = (const float*)d_in[7];
  const float* W2f  = (const float*)d_in[8];
  const float* b2f  = (const float*)d_in[9];
  const int* batch  = (const int*)d_in[10];
  const int* eidx   = (const int*)d_in[11];

  char* ws = (char*)d_ws;
  __bf16* WTe  = (__bf16*)ws;                              // 3.1 MB
  __bf16* WTf  = (__bf16*)(ws + (size_t)FF * DIN * 2);     // 3.1 MB
  float* ePart = (float*)(ws + (size_t)2 * FF * DIN * 2);  // 1 MB: [4096][64]

  float* energy = (float*)d_out;   // [64]
  float* forces = energy + NG;     // [50000*3]

  zero_out<<<(out_size + 255) / 256, 256, 0, stream>>>(energy, out_size);
  w1_transpose_cvt<<<dim3(DIN / 32, FF / 32, 2), dim3(32, 8, 1), 0, stream>>>(
      W1e, W1f, WTe, WTf);
  edge_mlp_kernel<<<NE / 64, 256, LDS_TOTAL, stream>>>(
      x, pos, b1e, W2e, b2e, b1f, W2f, b2f, batch, eidx, WTe, WTf, forces, ePart);
  energy_reduce<<<NG, 256, 0, stream>>>(ePart, energy);

  (void)in_sizes; (void)n_in; (void)ws_size;
}
